// Attention_agent_50525995270754
// MI455X (gfx1250) — matrix operand order repaired and checked by reading, not compiled, not run on hardware
//
#include <hip/hip_runtime.h>
#include <hip/hip_bf16.h>

#define DEV __device__ __forceinline__

typedef __bf16 bf16_t;
typedef __attribute__((ext_vector_type(16))) __bf16 bf16x16;
typedef __attribute__((ext_vector_type(8)))  float  f32x8;

// ---- problem constants ----
constexpr int Bc = 16, Cc = 256, Hc = 64, Wc = 64;
constexpr int NHc = 8, AGc = 64, KDc = 32;
constexpr int Nc = Hc * Wc;        // 4096
constexpr int O3 = 3 * Cc;         // 768
constexpr int KCH = 8;             // split-K chunks for k_attn
constexpr float SCALE = 0.17677669529663687f;  // KD^-0.5

// ---- workspace layout (bytes) ----
constexpr size_t OFF_QKV   = 0;                                    // bf16 [B][768][N]
constexpr size_t SZ_QKV    = (size_t)Bc * O3 * Nc * 2;
constexpr size_t OFF_WQKV  = OFF_QKV + SZ_QKV;                     // bf16 [768][256]
constexpr size_t OFF_BQKV  = OFF_WQKV + (size_t)O3 * Cc * 2;       // f32  [768]
constexpr size_t OFF_WPROJ = OFF_BQKV + (size_t)O3 * 4;            // bf16 [256][256]
constexpr size_t OFF_BPROJ = OFF_WPROJ + (size_t)Cc * Cc * 2;      // f32  [256]
constexpr size_t OFF_A     = OFF_BPROJ + (size_t)Cc * 4;           // bf16 [B][NH][KD][AG]
constexpr size_t OFF_PB    = OFF_A + (size_t)Bc * NHc * KDc * AGc * 2; // f32 [NH][AG][N]
constexpr size_t OFF_AB    = OFF_PB + (size_t)NHc * AGc * Nc * 4;  // f32 [NH][N][AG]
constexpr size_t OFF_SC    = OFF_AB + (size_t)NHc * Nc * AGc * 4;  // f32 scores [B][NH][AG][N]
constexpr size_t SZ_SC     = (size_t)Bc * NHc * AGc * Nc * 4;
constexpr size_t OFF_ATTNK = OFF_SC + SZ_SC;                       // bf16 [B][NH][AG][N]
constexpr size_t OFF_ATTN  = OFF_ATTNK + (size_t)Bc * NHc * AGc * Nc * 2; // bf16 [B][NH][AG][KD]
constexpr size_t OFF_ATTNP = OFF_ATTN + (size_t)Bc * NHc * AGc * KDc * 2; // f32 [KCH][B*NH][AG][KD]
constexpr size_t OFF_QATTN = OFF_SC;                               // bf16 [B][NH][N][AG] (reuse scores lo-half)
constexpr size_t OFF_OUT1  = OFF_SC + (size_t)Bc * NHc * Nc * AGc * 2;    // bf16 [B][N][C] (scores hi-half)

// ---- WMMA fragment helpers (wave32, 16x16x32 bf16) ----
DEV bf16x16 frag_a(const bf16_t* tile, int ld, int mbase, int kbase) {
  int lane = threadIdx.x & 31;
  const bf16_t* row = tile + (size_t)(mbase + (lane & 15)) * ld + kbase;
  int h2 = lane >> 4;
  union { bf16x16 v; uint4 q[2]; } u;
  u.q[0] = *(const uint4*)(row + 8 * h2);
  u.q[1] = *(const uint4*)(row + 16 + 8 * h2);
  return u.v;
}
DEV bf16x16 frag_b(const bf16_t* tile, int ld, int nbase, int kbase) {
  int lane = threadIdx.x & 31;
  const bf16_t* row = tile + (size_t)(nbase + (lane & 15)) * ld + kbase;
  int h2 = lane >> 4;
  union { bf16x16 v; uint4 q[2]; } u;
  u.q[0] = *(const uint4*)(row + 8 * h2);
  u.q[1] = *(const uint4*)(row + 16 + 8 * h2);
  return u.v;
}
DEV f32x8 wmma_bf16(bf16x16 a, bf16x16 b, f32x8 c) {
  return __builtin_amdgcn_wmma_f32_16x16x32_bf16(false, a, false, b, (short)0, c,
                                                 false, false);
}

// ---- K0: fold BN into GEMM weights: W'[o][c]=W*g*rsqrt(v+eps); b'=b-m*inv ----
__global__ void k_fold(const float* __restrict__ w, const float* __restrict__ bn,
                       int O, int Cin, bf16_t* __restrict__ wout, float* __restrict__ bout) {
  int idx = blockIdx.x * blockDim.x + threadIdx.x;
  if (idx >= O * Cin) return;
  int o = idx / Cin;
  float g = bn[0 * O + o], bb = bn[1 * O + o], m = bn[2 * O + o], v = bn[3 * O + o];
  float inv = g * rsqrtf(v + 1e-5f);
  wout[idx] = (bf16_t)(w[idx] * inv);
  if ((idx % Cin) == 0) bout[o] = bb - m * inv;
}

// ---- bilinear 7x7 -> 64x64 (half-pixel centers, edge clamp) ----
DEV float bilin7(const float* __restrict__ p, int y, int x) {
  float fy = (y + 0.5f) * (7.0f / 64.0f) - 0.5f;
  float fx = (x + 0.5f) * (7.0f / 64.0f) - 0.5f;
  fy = fminf(fmaxf(fy, 0.0f), 6.0f);
  fx = fminf(fmaxf(fx, 0.0f), 6.0f);
  int iy = (int)fy, ix = (int)fx;
  float ty = fy - iy, tx = fx - ix;
  int iy1 = min(iy + 1, 6), ix1 = min(ix + 1, 6);
  float v00 = p[iy * 7 + ix], v01 = p[iy * 7 + ix1];
  float v10 = p[iy1 * 7 + ix], v11 = p[iy1 * 7 + ix1];
  return (v00 * (1.f - tx) + v01 * tx) * (1.f - ty) + (v10 * (1.f - tx) + v11 * tx) * ty;
}

__global__ void k_pb(const float* __restrict__ an, const float* __restrict__ ah,
                     const float* __restrict__ aw, float* __restrict__ pb) {
  int idx = blockIdx.x * blockDim.x + threadIdx.x;
  if (idx >= NHc * AGc * Nc) return;
  int n = idx % Nc, hg = idx / Nc;
  int y = n >> 6, x = n & 63;
  float v = bilin7(an + (size_t)hg * 49, y, x);
  v += ah[(size_t)hg * Hc + y] + aw[(size_t)hg * Wc + x];
  pb[idx] = v;
}

__global__ void k_ab(const float* __restrict__ na, const float* __restrict__ ha,
                     const float* __restrict__ wa, float* __restrict__ ab) {
  int idx = blockIdx.x * blockDim.x + threadIdx.x;
  if (idx >= NHc * Nc * AGc) return;
  int g = idx % AGc, hn = idx / AGc;
  int n = hn % Nc, h = hn / Nc;
  int y = n >> 6, x = n & 63;
  float v = bilin7(na + ((size_t)h * AGc + g) * 49, y, x);
  v += ha[((size_t)h * Hc + y) * AGc + g] + wa[((size_t)h * Wc + x) * AGc + g];
  ab[idx] = v;
}

// ---- K1: qkv = BN-folded(W[768x256] @ X[256xN]) in bf16 ----
__global__ void __launch_bounds__(256)
k_qkv_gemm(const float* __restrict__ x, const bf16_t* __restrict__ wq,
           const float* __restrict__ bq, bf16_t* __restrict__ qkv) {
  __shared__ __align__(16) bf16_t As[64 * 32];  // [o][c]
  __shared__ __align__(16) bf16_t Bt[64 * 32];  // [n][c]
  const int n0 = blockIdx.x * 64, o0 = blockIdx.y * 64, b = blockIdx.z;
  const int tid = threadIdx.x, lane = tid & 31, wave = tid >> 5;
  const int wm = wave & 3, wn = wave >> 2;
  f32x8 acc0 = {}, acc1 = {};
  for (int kk = 0; kk < Cc; kk += 32) {
    { int r = tid >> 2, cg = (tid & 3) * 8;
      *(uint4*)(As + r * 32 + cg) = *(const uint4*)(wq + (size_t)(o0 + r) * Cc + kk + cg); }
    { // pair-pack fp32 x -> bf16 Bt[n][c]; 2x float4 loads, 4x b32 stores
      int cp = tid >> 4;              // c-pair index 0..15 (c = 2*cp)
      int ng = (tid & 15) * 4;        // n offset, 16 lanes sweep 64 contiguous n
      const float* xp0 = x + ((size_t)b * Cc + kk + 2 * cp) * Nc + n0 + ng;
      float4 f0 = *(const float4*)xp0;
      float4 f1 = *(const float4*)(xp0 + Nc);
      union { bf16_t h[2]; unsigned u; } p0, p1, p2, p3;
      p0.h[0] = (bf16_t)f0.x; p0.h[1] = (bf16_t)f1.x;
      p1.h[0] = (bf16_t)f0.y; p1.h[1] = (bf16_t)f1.y;
      p2.h[0] = (bf16_t)f0.z; p2.h[1] = (bf16_t)f1.z;
      p3.h[0] = (bf16_t)f0.w; p3.h[1] = (bf16_t)f1.w;
      unsigned* brow = (unsigned*)Bt;  // row = 32 bf16 = 16 dwords
      brow[(ng + 0) * 16 + cp] = p0.u;
      brow[(ng + 1) * 16 + cp] = p1.u;
      brow[(ng + 2) * 16 + cp] = p2.u;
      brow[(ng + 3) * 16 + cp] = p3.u; }
    __syncthreads();
    bf16x16 af = frag_a(As, 32, wm * 16, 0);
    acc0 = wmma_bf16(af, frag_b(Bt, 32, wn * 32, 0), acc0);
    acc1 = wmma_bf16(af, frag_b(Bt, 32, wn * 32 + 16, 0), acc1);
    __syncthreads();
  }
  int h2 = lane >> 4, nl = lane & 15;
  for (int r = 0; r < 8; ++r) {
    int o = o0 + wm * 16 + r + 8 * h2;
    float bias = bq[o];
    int n_a = n0 + wn * 32 + nl;
    qkv[((size_t)b * O3 + o) * Nc + n_a]      = (bf16_t)(acc0[r] + bias);
    qkv[((size_t)b * O3 + o) * Nc + n_a + 16] = (bf16_t)(acc1[r] + bias);
  }
}

// ---- K2: agent tokens via 8x8 pooling of q (with the (H,W,C) reinterpretation) ----
// q packed channel cq = h*32+kd lives at qkv channel o = (cq>>5)*96 + (cq&31)
__global__ void k_agent_pool(const bf16_t* __restrict__ qkv, bf16_t* __restrict__ a_out) {
  int idx = blockIdx.x * blockDim.x + threadIdx.x;
  if (idx >= Bc * Cc * AGc) return;
  int b = idx / (Cc * AGc);
  int rem = idx % (Cc * AGc);
  int c = rem / AGc, g = rem % AGc;
  int p1 = g >> 3, p2 = g & 7;
  const bf16_t* base = qkv + (size_t)b * O3 * Nc;
  float s = 0.f;
  for (int j = 0; j < 8; ++j)
    for (int l = 0; l < 8; ++l) {
      int y2 = p1 * 8 + j, x2 = p2 * 8 + l;
      int f = (y2 * Wc + x2) * Cc + c;          // flat index into packed q [C*N]
      int cq = f >> 12, n = f & (Nc - 1);       // N = 4096
      int o = (cq >> 5) * 96 + (cq & 31);       // q slice of qkv
      s += (float)base[(size_t)o * Nc + n];
    }
  int h = c >> 5, kd = c & 31;
  a_out[(((size_t)b * NHc + h) * KDc + kd) * AGc + g] = (bf16_t)(s * (1.0f / 64.0f));
}

// ---- K4a: scores[b,h,g,n] = scale * a^T k + pb (M=64,K=32,N=4096) ----
__global__ void __launch_bounds__(256)
k_scores(const bf16_t* __restrict__ qkv, const bf16_t* __restrict__ amat,
         const float* __restrict__ pb, float* __restrict__ scores) {
  __shared__ __align__(16) bf16_t Ast[64 * 32];  // [g][kd]
  __shared__ __align__(16) bf16_t Bt[64 * 32];   // [n][kd]
  int n0 = blockIdx.x * 64;
  int bh = blockIdx.y, b = bh / NHc, h = bh % NHc;
  int tid = threadIdx.x, lane = tid & 31, wave = tid >> 5;
  int wm = wave & 3, wn = wave >> 2;
  { int g = tid >> 2, kg = (tid & 3) * 8;
    const bf16_t* ap = amat + (size_t)bh * KDc * AGc;
    for (int i = 0; i < 8; ++i) Ast[g * 32 + kg + i] = ap[(size_t)(kg + i) * AGc + g]; }
  { int kd = tid >> 3, ng = (tid & 7) * 8;
    const bf16_t* kp = qkv + ((size_t)b * O3 + h * 96 + 32 + kd) * Nc + n0 + ng;
    bf16_t tmp[8]; *(uint4*)tmp = *(const uint4*)kp;
    for (int i = 0; i < 8; ++i) Bt[(ng + i) * 32 + kd] = tmp[i]; }
  __syncthreads();
  f32x8 acc0 = {}, acc1 = {};
  bf16x16 af = frag_a(Ast, 32, wm * 16, 0);
  acc0 = wmma_bf16(af, frag_b(Bt, 32, wn * 32, 0), acc0);
  acc1 = wmma_bf16(af, frag_b(Bt, 32, wn * 32 + 16, 0), acc1);
  int h2 = lane >> 4, nl = lane & 15;
  for (int r = 0; r < 8; ++r) {
    int g = wm * 16 + r + 8 * h2;
    int n_a = n0 + wn * 32 + nl;
    size_t pbase = ((size_t)h * AGc + g) * Nc;
    size_t sbase = ((size_t)bh * AGc + g) * Nc;
    scores[sbase + n_a]      = acc0[r] * SCALE + pb[pbase + n_a];
    scores[sbase + n_a + 16] = acc1[r] * SCALE + pb[pbase + n_a + 16];
  }
}

// ---- K4b: row softmax over N=4096, write bf16 attn_k ----
__global__ void __launch_bounds__(256)
k_softmax_rows(const float* __restrict__ scores, bf16_t* __restrict__ attnk) {
  __shared__ float red[256];
  int row = blockIdx.x;
  const float* sp = scores + (size_t)row * Nc;
  int t = threadIdx.x;
  float vals[16];
  float mx = -3.4e38f;
  for (int i = 0; i < 16; ++i) { vals[i] = sp[t + i * 256]; mx = fmaxf(mx, vals[i]); }
  red[t] = mx; __syncthreads();
  for (int s = 128; s > 0; s >>= 1) { if (t < s) red[t] = fmaxf(red[t], red[t + s]); __syncthreads(); }
  mx = red[0]; __syncthreads();
  float sum = 0.f;
  for (int i = 0; i < 16; ++i) { vals[i] = __expf(vals[i] - mx); sum += vals[i]; }
  red[t] = sum; __syncthreads();
  for (int s = 128; s > 0; s >>= 1) { if (t < s) red[t] += red[t + s]; __syncthreads(); }
  float inv = 1.0f / red[0];
  bf16_t* op = attnk + (size_t)row * Nc;
  for (int i = 0; i < 16; ++i) op[t + i * 256] = (bf16_t)(vals[i] * inv);
}

// ---- K5: attn = attn_k @ v^T (M=64,K=4096,N'=32), split-K into KCH chunks ----
__global__ void __launch_bounds__(256)
k_attn(const bf16_t* __restrict__ attnk, const bf16_t* __restrict__ qkv,
       float* __restrict__ attn_part) {
  __shared__ __align__(16) bf16_t As[64 * 32];  // [g][nchunk]
  __shared__ __align__(16) bf16_t Bt[32 * 32];  // [kd][nchunk]
  int kc = blockIdx.x;                           // K chunk
  int bh = blockIdx.y, b = bh / NHc, h = bh % NHc;
  int tid = threadIdx.x, lane = tid & 31, wave = tid >> 5;
  int wm = wave & 3, wn = wave >> 2;
  f32x8 acc = {};
  const bf16_t* ak = attnk + (size_t)bh * AGc * Nc;
  const bf16_t* vm = qkv + ((size_t)b * O3 + h * 96 + 64) * Nc;
  const int nkEnd = (kc + 1) * (Nc / KCH);
  for (int nk = kc * (Nc / KCH); nk < nkEnd; nk += 32) {
    { int g = tid >> 2, ng = (tid & 3) * 8;
      *(uint4*)(As + g * 32 + ng) = *(const uint4*)(ak + (size_t)g * Nc + nk + ng); }
    if (tid < 128) { int kd = tid >> 2, ng = (tid & 3) * 8;
      *(uint4*)(Bt + kd * 32 + ng) = *(const uint4*)(vm + (size_t)kd * Nc + nk + ng); }
    __syncthreads();
    acc = wmma_bf16(frag_a(As, 32, wm * 16, 0), frag_b(Bt, 32, wn * 16, 0), acc);
    __syncthreads();
  }
  int h2 = lane >> 4, nl = lane & 15;
  for (int r = 0; r < 8; ++r) {
    int g = wm * 16 + r + 8 * h2;
    int kd = wn * 16 + nl;
    attn_part[(((size_t)kc * (Bc * NHc) + bh) * AGc + g) * KDc + kd] = acc[r];
  }
}

__global__ void k_attn_reduce(const float* __restrict__ attn_part, bf16_t* __restrict__ attn) {
  int idx = blockIdx.x * blockDim.x + threadIdx.x;
  if (idx >= Bc * NHc * AGc * KDc) return;
  float s = 0.f;
  for (int kc = 0; kc < KCH; ++kc)
    s += attn_part[(size_t)kc * (Bc * NHc * AGc * KDc) + idx];
  attn[idx] = (bf16_t)s;
}

// ---- K6: q_attn = softmax_g(scale * q^T a + ab), fused (row = 64 cols) ----
__global__ void __launch_bounds__(256)
k_qattn(const bf16_t* __restrict__ qkv, const bf16_t* __restrict__ amat,
        const float* __restrict__ ab, bf16_t* __restrict__ qattn) {
  __shared__ __align__(16) bf16_t At[64 * 32];  // [n][kd]
  __shared__ __align__(16) bf16_t Bt[64 * 32];  // [g][kd]
  __shared__ float Sf[64 * 64];
  int n0 = blockIdx.x * 64;
  int bh = blockIdx.y, b = bh / NHc, h = bh % NHc;
  int tid = threadIdx.x, lane = tid & 31, wave = tid >> 5;
  int wm = wave & 3, wn = wave >> 2;
  { int kd = tid >> 3, ng = (tid & 7) * 8;
    const bf16_t* qp = qkv + ((size_t)b * O3 + h * 96 + kd) * Nc + n0 + ng;
    bf16_t tmp[8]; *(uint4*)tmp = *(const uint4*)qp;
    for (int i = 0; i < 8; ++i) At[(ng + i) * 32 + kd] = tmp[i]; }
  { int g = tid >> 2, kg = (tid & 3) * 8;
    const bf16_t* ap = amat + (size_t)bh * KDc * AGc;
    for (int i = 0; i < 8; ++i) Bt[g * 32 + kg + i] = ap[(size_t)(kg + i) * AGc + g]; }
  __syncthreads();
  f32x8 acc0 = {}, acc1 = {};
  bf16x16 af = frag_a(At, 32, wm * 16, 0);
  acc0 = wmma_bf16(af, frag_b(Bt, 32, wn * 32, 0), acc0);
  acc1 = wmma_bf16(af, frag_b(Bt, 32, wn * 32 + 16, 0), acc1);
  int h2 = lane >> 4, nl = lane & 15;
  for (int r = 0; r < 8; ++r) {
    int nr = wm * 16 + r + 8 * h2;
    int n_a = n0 + nr;
    int g0 = wn * 32 + nl;
    Sf[nr * 64 + g0]      = acc0[r] * SCALE + ab[((size_t)h * Nc + n_a) * AGc + g0];
    Sf[nr * 64 + g0 + 16] = acc1[r] * SCALE + ab[((size_t)h * Nc + n_a) * AGc + g0 + 16];
  }
  __syncthreads();
  if (tid < 64) {
    float mx = -3.4e38f;
    for (int g = 0; g < 64; ++g) mx = fmaxf(mx, Sf[tid * 64 + g]);
    float sum = 0.f;
    for (int g = 0; g < 64; ++g) { float e = __expf(Sf[tid * 64 + g] - mx); Sf[tid * 64 + g] = e; sum += e; }
    float inv = 1.f / sum;
    bf16_t* op = qattn + ((size_t)bh * Nc + n0 + tid) * AGc;
    for (int g = 0; g < 64; ++g) op[g] = (bf16_t)(Sf[tid * 64 + g] * inv);
  }
}

// ---- K7: out1[b,n,c] = q_attn @ attn  (M=4096,K=64,N'=32 per head) ----
__global__ void __launch_bounds__(256)
k_out1(const bf16_t* __restrict__ qattn, const bf16_t* __restrict__ attn,
       bf16_t* __restrict__ out1) {
  __shared__ __align__(16) bf16_t At[64 * 64];  // [n][g]
  __shared__ __align__(16) bf16_t Bt[32 * 64];  // [kd][g]
  int n0 = blockIdx.x * 64;
  int bh = blockIdx.y, b = bh / NHc, h = bh % NHc;
  int tid = threadIdx.x, lane = tid & 31, wave = tid >> 5;
  int wm = wave & 3, wn = wave >> 2;
  { int n = tid >> 2, gg = (tid & 3) * 16;
    const bf16_t* qp = qattn + ((size_t)bh * Nc + n0 + n) * AGc + gg;
    *(uint4*)(At + n * 64 + gg)     = *(const uint4*)qp;
    *(uint4*)(At + n * 64 + gg + 8) = *(const uint4*)(qp + 8); }
  { int kd = tid >> 3, gg = (tid & 7) * 8;
    const bf16_t* ap = attn + (size_t)bh * AGc * KDc;
    for (int i = 0; i < 8; ++i) Bt[kd * 64 + gg + i] = ap[(size_t)(gg + i) * KDc + kd]; }
  __syncthreads();
  f32x8 acc = {};
  for (int kb = 0; kb < 64; kb += 32)
    acc = wmma_bf16(frag_a(At, 64, wm * 16, kb), frag_b(Bt, 64, wn * 16, kb), acc);
  int h2 = lane >> 4, nl = lane & 15;
  for (int r = 0; r < 8; ++r) {
    int n_a = n0 + wm * 16 + r + 8 * h2;
    int c = h * KDc + wn * 16 + nl;
    out1[((size_t)b * Nc + n_a) * Cc + c] = (bf16_t)acc[r];
  }
}

// ---- K8: depthwise 3x3 conv on v + pe BN, added into out1 (c-major: coalesced RMW) ----
__global__ void k_pe_add(const bf16_t* __restrict__ qkv, const float* __restrict__ pw,
                         const float* __restrict__ pebn, bf16_t* __restrict__ out1) {
  size_t idx = (size_t)blockIdx.x * blockDim.x + threadIdx.x;
  if (idx >= (size_t)Bc * Cc * Nc) return;
  int c = (int)(idx % Cc);             // lanes sweep c -> out1 RMW contiguous
  size_t bn = idx / Cc;
  int n = (int)(bn % Nc);
  int b = (int)(bn / Nc);
  int y = n >> 6, x = n & 63;
  int o = (c >> 5) * 96 + 64 + (c & 31);
  const bf16_t* vm = qkv + ((size_t)b * O3 + o) * Nc;
  float s = 0.f;
  for (int dy = -1; dy <= 1; ++dy) {
    int yy = y + dy; if (yy < 0 || yy >= Hc) continue;
    for (int dx = -1; dx <= 1; ++dx) {
      int xx = x + dx; if (xx < 0 || xx >= Wc) continue;
      s += (float)vm[yy * Wc + xx] * pw[c * 9 + (dy + 1) * 3 + (dx + 1)];
    }
  }
  float g = pebn[0 * Cc + c], bb = pebn[1 * Cc + c];
  float m = pebn[2 * Cc + c], v = pebn[3 * Cc + c];
  float inv = g * rsqrtf(v + 1e-5f);
  float val = (s - m) * inv + bb;
  out1[idx] = (bf16_t)((float)out1[idx] + val);   // idx == ((b*N+n)*C + c)
}

// ---- K9: final projection GEMM + BN -> fp32 output ----
__global__ void __launch_bounds__(256)
k_proj(const bf16_t* __restrict__ out1, const bf16_t* __restrict__ wp,
       const float* __restrict__ bp, float* __restrict__ out) {
  __shared__ __align__(16) bf16_t As[64 * 32];  // [o][c]
  __shared__ __align__(16) bf16_t Bt[64 * 32];  // [n][c]
  int n0 = blockIdx.x * 64, o0 = blockIdx.y * 64, b = blockIdx.z;
  int tid = threadIdx.x, lane = tid & 31, wave = tid >> 5;
  int wm = wave & 3, wn = wave >> 2;
  f32x8 acc0 = {}, acc1 = {};
  for (int kk = 0; kk < Cc; kk += 32) {
    { int r = tid >> 2, cg = (tid & 3) * 8;
      *(uint4*)(As + r * 32 + cg) = *(const uint4*)(wp + (size_t)(o0 + r) * Cc + kk + cg); }
    { int n = tid >> 2, cg = (tid & 3) * 8;
      *(uint4*)(Bt + n * 32 + cg) =
          *(const uint4*)(out1 + ((size_t)b * Nc + n0 + n) * Cc + kk + cg); }
    __syncthreads();
    bf16x16 af = frag_a(As, 32, wm * 16, 0);
    acc0 = wmma_bf16(af, frag_b(Bt, 32, wn * 32, 0), acc0);
    acc1 = wmma_bf16(af, frag_b(Bt, 32, wn * 32 + 16, 0), acc1);
    __syncthreads();
  }
  int h2 = lane >> 4, nl = lane & 15;
  for (int r = 0; r < 8; ++r) {
    int o = o0 + wm * 16 + r + 8 * h2;
    float bias = bp[o];
    int n_a = n0 + wn * 32 + nl;
    out[((size_t)b * Cc + o) * Nc + n_a]      = acc0[r] + bias;
    out[((size_t)b * Cc + o) * Nc + n_a + 16] = acc1[r] + bias;
  }
}

extern "C" void kernel_launch(void* const* d_in, const int* in_sizes, int n_in,
                              void* d_out, int out_size, void* d_ws, size_t ws_size,
                              hipStream_t stream) {
  const float* x       = (const float*)d_in[0];
  const float* qkv_w   = (const float*)d_in[1];
  const float* qkv_bn  = (const float*)d_in[2];
  const float* pe_w    = (const float*)d_in[3];
  const float* pe_bn   = (const float*)d_in[4];
  const float* proj_w  = (const float*)d_in[5];
  const float* proj_bn = (const float*)d_in[6];
  const float* an_b    = (const float*)d_in[7];
  const float* na_b    = (const float*)d_in[8];
  const float* ah_b    = (const float*)d_in[9];
  const float* aw_b    = (const float*)d_in[10];
  const float* ha_b    = (const float*)d_in[11];
  const float* wa_b    = (const float*)d_in[12];
  float* out = (float*)d_out;
  char* ws = (char*)d_ws;

  bf16_t* qkv   = (bf16_t*)(ws + OFF_QKV);
  bf16_t* wqkv  = (bf16_t*)(ws + OFF_WQKV);
  float*  bqkv  = (float*)(ws + OFF_BQKV);
  bf16_t* wproj = (bf16_t*)(ws + OFF_WPROJ);
  float*  bproj = (float*)(ws + OFF_BPROJ);
  bf16_t* amat  = (bf16_t*)(ws + OFF_A);
  float*  pb    = (float*)(ws + OFF_PB);
  float*  ab    = (float*)(ws + OFF_AB);
  float*  sc    = (float*)(ws + OFF_SC);
  bf16_t* attnk = (bf16_t*)(ws + OFF_ATTNK);
  bf16_t* attn  = (bf16_t*)(ws + OFF_ATTN);
  float*  attnp = (float*)(ws + OFF_ATTNP);
  bf16_t* qattn = (bf16_t*)(ws + OFF_QATTN);
  bf16_t* out1  = (bf16_t*)(ws + OFF_OUT1);

  k_fold<<<(O3 * Cc + 255) / 256, 256, 0, stream>>>(qkv_w, qkv_bn, O3, Cc, wqkv, bqkv);
  k_fold<<<(Cc * Cc + 255) / 256, 256, 0, stream>>>(proj_w, proj_bn, Cc, Cc, wproj, bproj);
  k_pb<<<(NHc * AGc * Nc + 255) / 256, 256, 0, stream>>>(an_b, ah_b, aw_b, pb);
  k_ab<<<(NHc * Nc * AGc + 255) / 256, 256, 0, stream>>>(na_b, ha_b, wa_b, ab);
  k_qkv_gemm<<<dim3(Nc / 64, O3 / 64, Bc), 256, 0, stream>>>(x, wqkv, bqkv, qkv);
  k_agent_pool<<<(Bc * Cc * AGc + 255) / 256, 256, 0, stream>>>(qkv, amat);
  k_scores<<<dim3(Nc / 64, Bc * NHc), 256, 0, stream>>>(qkv, amat, pb, sc);
  k_softmax_rows<<<Bc * NHc * AGc, 256, 0, stream>>>(sc, attnk);
  k_attn<<<dim3(KCH, Bc * NHc), 256, 0, stream>>>(attnk, qkv, attnp);
  k_attn_reduce<<<(Bc * NHc * AGc * KDc + 255) / 256, 256, 0, stream>>>(attnp, attn);
  k_qattn<<<dim3(Nc / 64, Bc * NHc), 256, 0, stream>>>(qkv, amat, ab, qattn);
  k_out1<<<dim3(Nc / 64, Bc * NHc), 256, 0, stream>>>(qattn, attn, out1);
  k_pe_add<<<(int)(((size_t)Bc * Cc * Nc + 255) / 256), 256, 0, stream>>>(qkv, pe_w, pe_bn, out1);
  k_proj<<<dim3(Nc / 64, Cc / 64, Bc), 256, 0, stream>>>(out1, wproj, bproj, out);
}
